// noflayer_57080115364255
// MI455X (gfx1250) — matrix operand order repaired and checked by reading, not compiled, not run on hardware
//
#include <hip/hip_runtime.h>
#include <hip/hip_bf16.h>
#include <math.h>

typedef __attribute__((ext_vector_type(16))) _Float16 v16h;
typedef __attribute__((ext_vector_type(8)))  _Float16 v8h;
typedef __attribute__((ext_vector_type(8)))  float    v8f;

#define NN 8192
#define FF 128
#define KT (NN / 32)   // 256 K-tiles of 32
#define CT (FF / 16)   // 8 column tiles of 16
#define SPLITK 8       // K-splits for the WMMA GEMM (occupancy)
#define KT_PER_S (KT / SPLITK)   // 32 K-tiles per split (even)

// ---------------- feat_1 / feat_2 GEMV: one wave per row ----------------
__global__ void __launch_bounds__(256)
gemv_attn(const float* __restrict__ x, const float* __restrict__ a,
          float* __restrict__ f1, float* __restrict__ f2)
{
    const int wave = threadIdx.x >> 5;
    const int lane = threadIdx.x & 31;
    const int row  = blockIdx.x * 8 + wave;
    const float4 xv = *reinterpret_cast<const float4*>(x + (size_t)row * FF + lane * 4);
    const float4 a1 = *reinterpret_cast<const float4*>(a + lane * 4);
    const float4 a2 = *reinterpret_cast<const float4*>(a + FF + lane * 4);
    float s1 = xv.x*a1.x + xv.y*a1.y + xv.z*a1.z + xv.w*a1.w;
    float s2 = xv.x*a2.x + xv.y*a2.y + xv.z*a2.z + xv.w*a2.w;
#pragma unroll
    for (int m = 16; m >= 1; m >>= 1) {
        s1 += __shfl_xor(s1, m, 32);
        s2 += __shfl_xor(s2, m, 32);
    }
    if (lane == 0) { f1[row] = s1; f2[row] = s2; }
}

// ---- pack f32 [NN x FF] matrix into per-lane WMMA B fragments (f16) ----
// B fragment for v_wmma_f32_16x16x32_f16 (32x16 B):
//   lane l: col = (l&15), K = kt*32 + (l>>4)*8 + {0..7, 16..23}, 16 f16 per lane.
// Storage order: P[kt*CT*32 + ct*32 + lane].
__global__ void __launch_bounds__(256)
pack_b(const float* __restrict__ S, v16h* __restrict__ P)
{
    const int gid  = blockIdx.x * blockDim.x + threadIdx.x;  // 0 .. KT*CT*32-1
    const int lane = gid & 31;
    const int ct   = (gid >> 5) & (CT - 1);
    const int kt   = gid >> 8;
    const int col  = (ct << 4) + (lane & 15);
    const int k0   = (kt << 5) + ((lane >> 4) << 3);
    v16h v;
#pragma unroll
    for (int k = 0; k < 16; ++k)
        v[k] = (_Float16)S[(size_t)(k0 + (k < 8 ? k : k + 8)) * FF + col];
    P[gid] = v;
}

// ---------------- per-row masked softmax: one block per row ----------------
__global__ void __launch_bounds__(256)
attn_softmax(const float* __restrict__ adj, const float* __restrict__ f1,
             const float* __restrict__ f2, _Float16* __restrict__ U,
             float* __restrict__ rowsum)
{
    __shared__ float se[NN];
    __shared__ float red[8];
    const int i    = blockIdx.x;
    const int wave = threadIdx.x >> 5;
    const int lane = threadIdx.x & 31;
    const float fi = f1[i];
    const float* arow = adj + (size_t)i * NN;

    float vmax = -INFINITY;
    for (int j = threadIdx.x; j < NN; j += 256) {
        float e = fi + f2[j];
        e = (e > 0.f) ? e : 0.2f * e;                 // leaky_relu
        e = (arow[j] > 0.f) ? e : -9e15f;             // mask
        se[j] = e;
        vmax = fmaxf(vmax, e);
    }
#pragma unroll
    for (int m = 16; m >= 1; m >>= 1) vmax = fmaxf(vmax, __shfl_xor(vmax, m, 32));
    if (lane == 0) red[wave] = vmax;
    __syncthreads();
    float rmax = red[0];
#pragma unroll
    for (int w = 1; w < 8; ++w) rmax = fmaxf(rmax, red[w]);
    __syncthreads();

    float s = 0.f;
    for (int j = threadIdx.x; j < NN; j += 256) {
        float ex = __expf(se[j] - rmax);
        se[j] = ex;
        s += ex;
    }
#pragma unroll
    for (int m = 16; m >= 1; m >>= 1) s += __shfl_xor(s, m, 32);
    if (lane == 0) red[wave] = s;
    __syncthreads();
    float tot = 0.f;
#pragma unroll
    for (int w = 0; w < 8; ++w) tot += red[w];
    __syncthreads();

    const float inv = 1.f / tot;
    float rs = 0.f;
    _Float16* urow = U + (size_t)i * NN;
    for (int j = threadIdx.x; j < NN; j += 256) {
        float u = se[j] * inv;
        urow[j] = (_Float16)u;
        rs += 0.5f * u * u;
    }
#pragma unroll
    for (int m = 16; m >= 1; m >>= 1) rs += __shfl_xor(rs, m, 32);
    if (lane == 0) red[wave] = rs;
    __syncthreads();
    if (threadIdx.x == 0) {
        float t = 0.f;
#pragma unroll
        for (int w = 0; w < 8; ++w) t += red[w];
        rowsum[i] = t;
    }
}

// ------------------ WMMA GEMM helpers ------------------
__device__ __forceinline__ void load_bfrags(const v16h* __restrict__ base, int kt,
                                            int lane, v16h b[CT])
{
    const v16h* brow = base + (size_t)kt * (CT * 32) + lane;
#pragma unroll
    for (int ct = 0; ct < CT; ++ct) b[ct] = brow[ct * 32];
}

__device__ __forceinline__ v16h load_afrag(const _Float16* __restrict__ Urow,
                                           int kt, int khalf)
{
    const int kbase = (kt << 5) + (khalf << 3);
    const v8h a0 = *reinterpret_cast<const v8h*>(Urow + kbase);
    const v8h a1 = *reinterpret_cast<const v8h*>(Urow + kbase + 16);
    v16h a;
#pragma unroll
    for (int q = 0; q < 8; ++q) { a[q] = a0[q]; a[8 + q] = a1[q]; }
    return a;
}

__device__ __forceinline__ void wmma_step(const v16h& a, const v16h b[CT], v8f acc[CT])
{
#pragma unroll
    for (int ct = 0; ct < CT; ++ct) {
        acc[ct] = __builtin_amdgcn_wmma_f32_16x16x32_f16(
            false, a, false, b[ct], (short)0, acc[ct], false, false);
    }
}

// -------- split-K WMMA GEMM: Pp[s] = U[:, sK:(s+1)K] @ B[sK:(s+1)K, :] ----
// 4096 waves (512 row-strips x 8 K-splits). Both A and B fragments are
// double-buffered in registers: every operand a WMMA group consumes was issued
// one full stage earlier, with ~18 loads issued since, so the wait before the
// WMMAs is s_wait_loadcnt ~18 instead of 0 — a full K-step of matrix work
// overlaps a full K-step of memory latency. sched_barrier(0) keeps the
// scheduler from re-sinking the loads into the WMMA sequence.
__global__ void __launch_bounds__(128)
gemm_wmma_splitk(const _Float16* __restrict__ U, const v16h* __restrict__ Bp,
                 float* __restrict__ Pp)
{
    const int w     = (blockIdx.x << 2) | (threadIdx.x >> 5);   // 0..4095
    const int lane  = threadIdx.x & 31;
    const int s     = w & (SPLITK - 1);
    const int rt    = w >> 3;                                   // 0..511
    const int rowA  = (rt << 4) + (lane & 15);
    const int khalf = lane >> 4;     // 0: K 0-7 & 16-23 ; 1: K 8-15 & 24-31
    const _Float16* Urow = U + (size_t)rowA * NN;

    const v8f vzero = {0.f, 0.f, 0.f, 0.f, 0.f, 0.f, 0.f, 0.f};
    v8f acc[CT];
#pragma unroll
    for (int t = 0; t < CT; ++t) acc[t] = vzero;

    const int kt0 = s * KT_PER_S;
    v16h b0[CT], b1[CT];
    v16h aA, aB;
    load_bfrags(Bp, kt0, lane, b0);              // prologue: stage-0 operands
    aA = load_afrag(Urow, kt0, khalf);

    for (int kt = kt0; kt < kt0 + KT_PER_S; kt += 2) {
        // stage 0: issue kt+1 operands, then WMMA on (aA, b0) from last stage
        load_bfrags(Bp, kt + 1, lane, b1);
        aB = load_afrag(Urow, kt + 1, khalf);
        __builtin_prefetch((const void*)(Urow + ((kt + 4) << 5)), 0, 1);
        __builtin_amdgcn_sched_barrier(0);
        wmma_step(aA, b0, acc);
        __builtin_amdgcn_sched_barrier(0);
        // stage 1: issue kt+2 operands (B pad keeps this in-bounds; the tail
        // A-load lands in the next U row / pad and its value is never used)
        load_bfrags(Bp, kt + 2, lane, b0);
        aA = load_afrag(Urow, kt + 2, khalf);
        __builtin_amdgcn_sched_barrier(0);
        wmma_step(aB, b1, acc);
        __builtin_amdgcn_sched_barrier(0);
    }

    // C/D layout: VGPR v -> row (lane<16 ? v : v+8), col = lane&15
    float* Cs = Pp + (size_t)s * (NN * FF);
    const int colBase = lane & 15;
    const int rowOff  = (lane >> 4) * 8;
#pragma unroll
    for (int ct = 0; ct < CT; ++ct) {
#pragma unroll
        for (int v = 0; v < 8; ++v) {
            const int row = (rt << 4) + rowOff + v;
            Cs[(size_t)row * FF + (ct << 4) + colBase] = acc[ct][v];
        }
    }
}

// ---------------- lifting epilogues (fold split-K reduction in) ----------
__device__ inline float sigm(float t) { return 1.f / (1.f + __expf(-t)); }

__global__ void __launch_bounds__(256)
epilogue1(const float* __restrict__ Pp, const float* __restrict__ x,
          const float* __restrict__ rowsum, const float* __restrict__ temp,
          float* __restrict__ feat)
{
    const size_t idx = (size_t)blockIdx.x * blockDim.x + threadIdx.x;
    const int i = (int)(idx >> 7);
    float p = 0.f;
#pragma unroll
    for (int s = 0; s < SPLITK; ++s) p += Pp[(size_t)s * (NN * FF) + idx];
    const float c2 = sigm(temp[2]);
    const float xv = x[idx];
    feat[idx] = p - xv * rowsum[i] + (1.f - c2) * xv;
}

__global__ void __launch_bounds__(256)
epilogue2(const float* __restrict__ Pp, const float* __restrict__ feat,
          const float* __restrict__ rowsum, const float* __restrict__ temp,
          const float* __restrict__ h0, float* __restrict__ out)
{
    const size_t idx = (size_t)blockIdx.x * blockDim.x + threadIdx.x;
    const int i = (int)(idx >> 7);
    float p2 = 0.f;
#pragma unroll
    for (int s = 0; s < SPLITK; ++s) p2 += Pp[(size_t)s * (NN * FF) + idx];
    const float c1 = sigm(temp[1]);
    const float c3 = sigm(temp[3]);
    const float odd = feat[idx] - p2;
    const float fp = c1 * c3 * p2 + (c1 * rowsum[i] + 1.f - c1) * odd;
    out[idx] = 0.2f * fp + 0.8f * h0[idx];
}

extern "C" void kernel_launch(void* const* d_in, const int* in_sizes, int n_in,
                              void* d_out, int out_size, void* d_ws, size_t ws_size,
                              hipStream_t stream)
{
    const float* x    = (const float*)d_in[0];   // [8192,128]
    const float* h0   = (const float*)d_in[1];   // [8192,128]
    const float* adj  = (const float*)d_in[2];   // [8192,8192]
    const float* a    = (const float*)d_in[3];   // [256]
    const float* temp = (const float*)d_in[4];   // [5]
    // d_in[5], d_in[6] (lamda, l) feed only the unused beta term.

    // Packed-B regions padded by 2 K-tiles (16 KB) so the steady-state
    // double-buffer prefetch of kt+2 never reads out of bounds.
    char* ws = (char*)d_ws;
    _Float16* U    = (_Float16*)(ws);                      // 128 MB f16 U
    v16h*     Xp   = (v16h*)(ws + 134217728ULL);           // 2 MB + 16 KB pad
    v16h*     Fp   = (v16h*)(ws + 136331264ULL);           // 2 MB + 16 KB pad
    float*    Pp   = (float*)(ws + 138444800ULL);          // 32 MB split-K partials
    float*    feat = (float*)(ws + 171999232ULL);          // 4 MB
    float*    f1   = (float*)(ws + 176193536ULL);          // 32 KB
    float*    f2   = (float*)(ws + 176226304ULL);          // 32 KB
    float*    rsum = (float*)(ws + 176259072ULL);          // 32 KB

    gemv_attn       <<<NN / 8, 256, 0, stream>>>(x, a, f1, f2);
    pack_b          <<<(KT * CT * 32) / 256, 256, 0, stream>>>(x, Xp);
    attn_softmax    <<<NN, 256, 0, stream>>>(adj, f1, f2, U, rsum);
    gemm_wmma_splitk<<<(NN / 16) * SPLITK / 4, 128, 0, stream>>>(U, Xp, Pp);
    epilogue1       <<<(NN * FF) / 256, 256, 0, stream>>>(Pp, x, rsum, temp, feat);
    pack_b          <<<(KT * CT * 32) / 256, 256, 0, stream>>>(feat, Fp);
    gemm_wmma_splitk<<<(NN / 16) * SPLITK / 4, 128, 0, stream>>>(U, Fp, Pp);
    epilogue2       <<<(NN * FF) / 256, 256, 0, stream>>>(Pp, feat, rsum, temp, h0,
                                                          (float*)d_out);
}
